// ReLSTM_18940805775611
// MI455X (gfx1250) — matrix operand order repaired and checked by reading, not compiled, not run on hardware
//
#include <hip/hip_runtime.h>
#include <hip/hip_bf16.h>

typedef __attribute__((ext_vector_type(16))) __bf16 v16bf;
typedef __attribute__((ext_vector_type(8)))  float  v8f;

#define HID      512
#define T_DIM    128
#define BT       16          // batch rows per workgroup (== WMMA M)
#define G4H      2048        // 4*HID gate columns
#define NTHREADS 512         // 16 waves of 32

// fast sigmoid / tanh via v_exp_f32 + v_rcp_f32 (TRANS ops, co-execute with WMMA)
__device__ __forceinline__ float sigf(float x) {
  float e = __builtin_amdgcn_exp2f(-1.4426950408889634f * x);
  return __builtin_amdgcn_rcpf(1.0f + e);
}
__device__ __forceinline__ float tanh_(float x) {
  float e = __builtin_amdgcn_exp2f(-2.8853900817779268f * x);
  return 2.0f * __builtin_amdgcn_rcpf(1.0f + e) - 1.0f;
}

// A fragment (16x32 bf16, M=batch rows) from LDS h-buffer, per ISA layout:
// lanes 0-15: M=lane, K = {k0..k0+7, k0+16..k0+23}; lanes 16-31: K shifted by 8.
__device__ __forceinline__ v16bf load_a(const __bf16* hs, int kt, int l15, int hi) {
  int base = l15 * HID + kt * 32 + hi * 8;
  union { v16bf v; uint4 q[2]; } u;
  u.q[0] = *(const uint4*)(hs + base);
  u.q[1] = *(const uint4*)(hs + base + 16);
  return u.v;
}

// B fragment (32x16 bf16, K x N) from row-major bf16 W [2048 x 512]:
// lane n(<16): column N=n holds K = {k0..k0+7, k0+16..k0+23}; lane n+16: K shifted by 8.
__device__ __forceinline__ v16bf load_b(const __bf16* W, int nt, int kt, int l15, int hi) {
  const __bf16* p = W + ((nt * 16 + l15) * HID + kt * 32 + hi * 8);
  union { v16bf v; uint4 q[2]; } u;
  u.q[0] = *(const uint4*)(p);
  u.q[1] = *(const uint4*)(p + 16);
  return u.v;
}

// acc[g*2+j] accumulates N-tile (g*32 + 2*wave + j); K = 512 in 16 steps of 32.
__device__ __forceinline__ void gemm8(v8f acc[8], const __bf16* hs, const __bf16* W,
                                      int wave, int l15, int hi) {
#pragma unroll 1
  for (int kt = 0; kt < 16; ++kt) {
    v16bf a = load_a(hs, kt, l15, hi);
#pragma unroll
    for (int g = 0; g < 4; ++g) {
#pragma unroll
      for (int j = 0; j < 2; ++j) {
        int nt = g * 32 + 2 * wave + j;
        v16bf b = load_b(W, nt, kt, l15, hi);
        acc[g * 2 + j] = __builtin_amdgcn_wmma_f32_16x16x32_bf16(
            false, a, false, b, (short)0, acc[g * 2 + j], false, false);
      }
    }
  }
}

__global__ void prep_bf16_kernel(const float* whh0, const float* wih1,
                                 const float* whh1, __bf16* ws) {
  int i = blockIdx.x * blockDim.x + threadIdx.x;
  const int n = G4H * HID;
  if (i < n) {
    ws[i]         = (__bf16)whh0[i];
    ws[i + n]     = (__bf16)wih1[i];
    ws[i + 2 * n] = (__bf16)whh1[i];
  }
}

__global__ __launch_bounds__(NTHREADS, 1)
void lstm_kernel(const float* __restrict__ x,
                 const float* __restrict__ w_ih0,
                 const float* __restrict__ b_ih0, const float* __restrict__ b_hh0,
                 const float* __restrict__ b_ih1, const float* __restrict__ b_hh1,
                 const float* __restrict__ w_lin, const float* __restrict__ b_lin,
                 const __bf16* __restrict__ wbf, float* __restrict__ out) {
  __shared__ __align__(16) __bf16 h1s[BT * HID];
  __shared__ __align__(16) __bf16 h2s[BT * HID];
  __shared__ float xtile[BT * T_DIM];
  __shared__ float wlin_s[HID];
  __shared__ float xcur[BT];
  __shared__ float ybuf[BT];

  const int tid  = threadIdx.x;
  const int lane = tid & 31;
  const int wave = tid >> 5;
  const int l15  = lane & 15;
  const int hi   = lane >> 4;
  const int bbase = blockIdx.x * BT;

  const __bf16* Whh0 = wbf;
  const __bf16* Wih1 = wbf + G4H * HID;
  const __bf16* Whh1 = wbf + 2 * G4H * HID;

  for (int i = tid; i < BT * HID; i += NTHREADS) {
    h1s[i] = (__bf16)0.0f;
    h2s[i] = (__bf16)0.0f;
  }
  for (int i = tid; i < BT * T_DIM; i += NTHREADS) {
    int row = i >> 7, tt = i & 127;
    xtile[i] = x[(bbase + row) * T_DIM + tt];
  }
  for (int i = tid; i < HID; i += NTHREADS) wlin_s[i] = w_lin[i];
  if (tid < BT) { ybuf[tid] = 0.0f; xcur[tid] = 0.0f; }

  // per-thread gate constants for the 8 owned (gate, tile) columns
  float b0c[4][2], wi0c[4][2], b1c[4][2];
#pragma unroll
  for (int g = 0; g < 4; ++g) {
#pragma unroll
    for (int j = 0; j < 2; ++j) {
      int col = (g * 32 + 2 * wave + j) * 16 + l15;
      b0c[g][j]  = b_ih0[col] + b_hh0[col];
      wi0c[g][j] = w_ih0[col];
      b1c[g][j]  = b_ih1[col] + b_hh1[col];
    }
  }
  const float blin = b_lin[0];
  float c1[16], c2[16];
#pragma unroll
  for (int k = 0; k < 16; ++k) { c1[k] = 0.0f; c2[k] = 0.0f; }

  __syncthreads();

  const v8f vzero = {0.f, 0.f, 0.f, 0.f, 0.f, 0.f, 0.f, 0.f};

  for (int t = 0; t < 2 * T_DIM - 1; ++t) {
    const bool phase1 = (t < T_DIM);

    // ---------------- layer 0: g0 = h1 @ Whh0^T (+ x*wih0 + b0) -------------
    v8f acc[8];
#pragma unroll
    for (int q = 0; q < 8; ++q) acc[q] = vzero;
    gemm8(acc, h1s, Whh0, wave, l15, hi);
    __syncthreads();  // all reads of old h1s complete

    float xv[8];
#pragma unroll
    for (int r = 0; r < 8; ++r) {
      int row = hi * 8 + r;  // C-matrix row M = r + 8*(lane>=16)
      xv[r] = phase1 ? xtile[row * T_DIM + t] : xcur[row];
    }
#pragma unroll
    for (int j = 0; j < 2; ++j) {
      int hcol = (2 * wave + j) * 16 + l15;
#pragma unroll
      for (int r = 0; r < 8; ++r) {
        float iv = acc[0 + j][r] + b0c[0][j] + xv[r] * wi0c[0][j];
        float fv = acc[2 + j][r] + b0c[1][j] + xv[r] * wi0c[1][j];
        float gv = acc[4 + j][r] + b0c[2][j] + xv[r] * wi0c[2][j];
        float ov = acc[6 + j][r] + b0c[3][j] + xv[r] * wi0c[3][j];
        float c  = sigf(fv) * c1[j * 8 + r] + sigf(iv) * tanh_(gv);
        c1[j * 8 + r] = c;
        float h = sigf(ov) * tanh_(c);
        h1s[(hi * 8 + r) * HID + hcol] = (__bf16)h;
      }
    }
    __syncthreads();  // new h1s visible

    // ---------------- layer 1: g1 = h1 @ Wih1^T + h2 @ Whh1^T + b1 ----------
#pragma unroll
    for (int q = 0; q < 8; ++q) acc[q] = vzero;
    gemm8(acc, h1s, Wih1, wave, l15, hi);
    gemm8(acc, h2s, Whh1, wave, l15, hi);
    __syncthreads();  // all reads of old h2s complete
#pragma unroll
    for (int j = 0; j < 2; ++j) {
      int hcol = (2 * wave + j) * 16 + l15;
#pragma unroll
      for (int r = 0; r < 8; ++r) {
        float iv = acc[0 + j][r] + b1c[0][j];
        float fv = acc[2 + j][r] + b1c[1][j];
        float gv = acc[4 + j][r] + b1c[2][j];
        float ov = acc[6 + j][r] + b1c[3][j];
        float c  = sigf(fv) * c2[j * 8 + r] + sigf(iv) * tanh_(gv);
        c2[j * 8 + r] = c;
        float h = sigf(ov) * tanh_(c);
        h2s[(hi * 8 + r) * HID + hcol] = (__bf16)h;
      }
    }
    __syncthreads();  // new h2s visible

    // ---------------- output head: y = h2 @ wlin^T + b ----------------------
    if (t >= T_DIM - 1) {
      if (tid < 256) {
        int row = tid >> 4, seg = tid & 15;
        float s = 0.0f;
#pragma unroll
        for (int k = 0; k < 32; ++k) {
          int cc = seg * 32 + k;
          s += (float)h2s[row * HID + cc] * wlin_s[cc];
        }
        atomicAdd(&ybuf[row], s);  // ds_add_f32
      }
      __syncthreads();
      if (tid < BT) {
        float yv = ybuf[tid] + blin;
        xcur[tid] = yv;       // next autoregressive input
        ybuf[tid] = 0.0f;     // reset before next use (next adds are past a barrier)
        out[(bbase + tid) * T_DIM + (t - (T_DIM - 1))] = yv;
      }
      // xcur read in next step happens after next step's first __syncthreads
    }
  }
}

extern "C" void kernel_launch(void* const* d_in, const int* in_sizes, int n_in,
                              void* d_out, int out_size, void* d_ws, size_t ws_size,
                              hipStream_t stream) {
  const float* x     = (const float*)d_in[0];
  // d_in[1]=x_time, d_in[2]=y_time: unused (as in reference)
  const float* w_ih0 = (const float*)d_in[3];
  const float* w_hh0 = (const float*)d_in[4];
  const float* b_ih0 = (const float*)d_in[5];
  const float* b_hh0 = (const float*)d_in[6];
  const float* w_ih1 = (const float*)d_in[7];
  const float* w_hh1 = (const float*)d_in[8];
  const float* b_ih1 = (const float*)d_in[9];
  const float* b_hh1 = (const float*)d_in[10];
  const float* w_lin = (const float*)d_in[11];
  const float* b_lin = (const float*)d_in[12];
  __bf16* wbf = (__bf16*)d_ws;  // 3 * 2048 * 512 * 2B = 6 MB of scratch

  const int n = G4H * HID;
  prep_bf16_kernel<<<(n + 255) / 256, 256, 0, stream>>>(w_hh0, w_ih1, w_hh1, wbf);

  const int nblocks = 512 / BT;  // B=512 -> 32 independent workgroups
  lstm_kernel<<<nblocks, NTHREADS, 0, stream>>>(
      x, w_ih0, b_ih0, b_hh0, b_ih1, b_hh1, w_lin, b_lin, wbf, (float*)d_out);
}
